// AADecoderBlock_4063039062796
// MI455X (gfx1250) — matrix operand order repaired and checked by reading, not compiled, not run on hardware
//
#include <hip/hip_runtime.h>
#include <hip/hip_bf16.h>

#define NRES 16384
#define KNBR 32

typedef __attribute__((ext_vector_type(16))) _Float16 v16h;
typedef __attribute__((ext_vector_type(8)))  _Float16 v8h;
typedef __attribute__((ext_vector_type(8)))  float    v8f;

__device__ __forceinline__ float geluf(float x) {
    const float kk = 0.7978845608028654f; // sqrt(2/pi)
    return 0.5f * x * (1.0f + tanhf(kk * (x + 0.044715f * x * x * x)));
}
__device__ __forceinline__ float wsum(float v) {
#pragma unroll
    for (int m = 16; m >= 1; m >>= 1) v += __shfl_xor(v, m, 32);
    return v;
}
__device__ __forceinline__ float wmaxf(float v) {
#pragma unroll
    for (int m = 16; m >= 1; m >>= 1) v = fmaxf(v, __shfl_xor(v, m, 32));
    return v;
}

// ---------------------------------------------------------------------------
// WMMA GEMM: out[rows x cols] = A_f16[rows x kdim] * (BT_f16[cols x kdim])^T
// Each wave computes a 16 x (16*NT) strip: one A fragment per K-step feeds NT
// independent v_wmma accumulator chains (breaks XDL RAW serialization, cuts
// A traffic NT-fold). Epilogue: +bias[col], optional gelu, optional
// +addSrc[row*cols+col], stores f32 and/or f16.
// ---------------------------------------------------------------------------
template <int NT>
__launch_bounds__(256)
__global__ void gemm_wmma_kernel(const _Float16* __restrict__ A,
                                 const _Float16* __restrict__ BT,
                                 const float* __restrict__ bias,
                                 const float* __restrict__ addSrc,
                                 float* __restrict__ outF32,
                                 _Float16* __restrict__ outF16,
                                 int rows, int kdim, int cols, int doGelu)
{
    const int lane = threadIdx.x & 31;
    const int wave = threadIdx.x >> 5;
    const int colStrips = cols / (16 * NT);
    const int rowTiles = rows >> 4;
    const int strip = blockIdx.x * 8 + wave;
    if (strip >= rowTiles * colStrips) return;   // wave-uniform exit: EXEC stays all-ones
    const int rt = strip / colStrips;
    const int cs = strip - rt * colStrips;
    const int lo = lane & 15;
    const int hi = lane >> 4;
    const size_t aRow = (size_t)(rt * 16 + lo) * kdim;
    size_t bRow[NT];
#pragma unroll
    for (int t = 0; t < NT; t++)
        bRow[t] = (size_t)((cs * NT + t) * 16 + lo) * kdim;

    v8f acc[NT];
#pragma unroll
    for (int t = 0; t < NT; t++) acc[t] = (v8f){};

    for (int kb = 0; kb < kdim; kb += 32) {
        // A 16x32 f16 fragment: lane holds row M=lo; elems 0-7 => K=kb+16*hi..,
        // elems 8-15 => K=kb+16*hi+8..
        v8h a0 = *(const v8h*)(A + aRow + kb + 16 * hi);
        v8h a1 = *(const v8h*)(A + aRow + kb + 16 * hi + 8);
        v16h av = __builtin_shufflevector(a0, a1, 0,1,2,3,4,5,6,7,8,9,10,11,12,13,14,15);
#pragma unroll
        for (int t = 0; t < NT; t++) {
            // B 32x16 f16 fragment: lane holds col N=lo, K=kb+16*hi..+15 contiguous
            v16h bv = *(const v16h*)(BT + bRow[t] + kb + 16 * hi);
            acc[t] = __builtin_amdgcn_wmma_f32_16x16x32_f16(false, av, false, bv,
                                                            (short)0, acc[t], false, false);
        }
    }

#pragma unroll
    for (int t = 0; t < NT; t++) {
        const int n = (cs * NT + t) * 16 + lo;
        const float bval = bias ? bias[n] : 0.0f;
#pragma unroll
        for (int r = 0; r < 8; r++) {
            const int m = rt * 16 + hi * 8 + r;
            float v = acc[t][r] + bval;
            if (doGelu) v = geluf(v);
            const size_t o = (size_t)m * cols + n;
            if (addSrc) v += addSrc[o];
            if (outF32) outF32[o] = v;
            if (outF16) outF16[o] = (_Float16)v;
        }
    }
}

// ---------------------------------------------------------------------------
// Weight prep: WT[n*Kpad + k] = (k < K) ? W[k*Ncols + n] : 0   (f32 -> f16)
// ---------------------------------------------------------------------------
__global__ void transpose_f16_kernel(const float* __restrict__ W, _Float16* __restrict__ WT,
                                     int Kd, int Nc, int Kpad)
{
    int t = blockIdx.x * 256 + threadIdx.x;
    if (t >= Nc * Kpad) return;
    int nidx = t / Kpad, k = t - nidx * Kpad;
    WT[t] = (k < Kd) ? (_Float16)W[(size_t)k * Nc + nidx] : (_Float16)0.0f;
}

// Wc = mlp_W2(128x64) @ Wb(64x8), bc = mlp_b2 @ Wb; WcT stored [16 x 128] f16 (rows 8..15 zero)
__global__ void make_wc_kernel(const float* __restrict__ W2, const float* __restrict__ Wb,
                               const float* __restrict__ b2,
                               _Float16* __restrict__ WcT, float* __restrict__ bc)
{
    int t = blockIdx.x * 256 + threadIdx.x;
    if (t >= 16 * 128) return;
    int h = t >> 7, k = t & 127;
    float s = 0.0f;
    if (h < 8) { for (int j = 0; j < 64; j++) s += W2[k * 64 + j] * Wb[j * 8 + h]; }
    WcT[h * 128 + k] = (_Float16)s;
    if (k == 0) {
        float tb = 0.0f;
        if (h < 8) { for (int j = 0; j < 64; j++) tb += b2[j] * Wb[j * 8 + h]; }
        bc[h] = tb;
    }
}

__global__ void f32_to_f16_kernel(const float* __restrict__ src, _Float16* __restrict__ dst, int count)
{
    int t = blockIdx.x * 256 + threadIdx.x;
    if (t < count) dst[t] = (_Float16)src[t];
}

// ---------------------------------------------------------------------------
// Frames + aa-features: one thread per node
// ---------------------------------------------------------------------------
__launch_bounds__(256)
__global__ void frames_kernel(const float* __restrict__ pos, const int* __restrict__ aa,
                              const float* __restrict__ W_aa, const float* __restrict__ lns,
                              const float* __restrict__ lnb,
                              float* __restrict__ R, float* __restrict__ ca,
                              _Float16* __restrict__ lp16, float* __restrict__ aaf, int Nn)
{
    int n = blockIdx.x * 256 + threadIdx.x;
    if (n >= Nn) return;
    const float* pp = pos + (size_t)n * 42;
    float Na0 = pp[0], Na1 = pp[1], Na2 = pp[2];
    float Ca0 = pp[3], Ca1 = pp[4], Ca2 = pp[5];
    float Cc0 = pp[6], Cc1 = pp[7], Cc2 = pp[8];
    float e10 = Cc0 - Ca0, e11 = Cc1 - Ca1, e12 = Cc2 - Ca2;
    float i1 = 1.0f / (sqrtf(e10 * e10 + e11 * e11 + e12 * e12) + 1e-8f);
    e10 *= i1; e11 *= i1; e12 *= i1;
    float u0 = Na0 - Ca0, u1 = Na1 - Ca1, u2 = Na2 - Ca2;
    float du = u0 * e10 + u1 * e11 + u2 * e12;
    u0 -= du * e10; u1 -= du * e11; u2 -= du * e12;
    float i2 = 1.0f / (sqrtf(u0 * u0 + u1 * u1 + u2 * u2) + 1e-8f);
    float e20 = u0 * i2, e21 = u1 * i2, e22 = u2 * i2;
    float e30 = e11 * e22 - e12 * e21;
    float e31 = e12 * e20 - e10 * e22;
    float e32 = e10 * e21 - e11 * e20;
    // R[c][d], columns e1,e2,e3
    float Rm[9] = { e10, e20, e30,  e11, e21, e31,  e12, e22, e32 };
#pragma unroll
    for (int q = 0; q < 9; q++) R[(size_t)n * 9 + q] = Rm[q];
    ca[(size_t)n * 3 + 0] = Ca0; ca[(size_t)n * 3 + 1] = Ca1; ca[(size_t)n * 3 + 2] = Ca2;
    // local_pos -> f16 padded to 64
    for (int a = 0; a < 14; a++) {
        float px = pp[a * 3 + 0] - Ca0, py = pp[a * 3 + 1] - Ca1, pz = pp[a * 3 + 2] - Ca2;
#pragma unroll
        for (int dd = 0; dd < 3; dd++) {
            float s = px * Rm[dd] + py * Rm[3 + dd] + pz * Rm[6 + dd];
            lp16[(size_t)n * 64 + a * 3 + dd] = (_Float16)s;
        }
    }
    for (int q = 42; q < 64; q++) lp16[(size_t)n * 64 + q] = (_Float16)0.0f;
    // aaf = LN(one_hot(aa) @ W_aa) = LN(row W_aa[aa])
    int ar = aa[n];
    float v[64], sum = 0.f, sq = 0.f;
    for (int c = 0; c < 64; c++) { float t = W_aa[ar * 64 + c]; v[c] = t; sum += t; sq += t * t; }
    float mu = sum * (1.0f / 64.0f);
    float var = sq * (1.0f / 64.0f) - mu * mu;
    float rs = rsqrtf(var + 1e-5f);
    for (int c = 0; c < 64; c++) aaf[(size_t)n * 64 + c] = (v[c] - mu) * rs * lns[c] + lnb[c];
}

// ---------------------------------------------------------------------------
// Pair features + LayerNorm -> f16 [chunk x 64]. One wave per pair, 2 cols/lane.
// ---------------------------------------------------------------------------
__launch_bounds__(256)
__global__ void pair_feat_kernel(const float* __restrict__ pos, const int* __restrict__ nbr,
                                 const int* __restrict__ resi, const int* __restrict__ chain,
                                 const int* __restrict__ batch,
                                 const float* __restrict__ R, const float* __restrict__ ca,
                                 const float* __restrict__ aaf,
                                 const float* __restrict__ Wrel, const float* __restrict__ Wdist,
                                 const float* __restrict__ Wdir, const float* __restrict__ Wrot,
                                 const float* __restrict__ Wpvec,
                                 const float* __restrict__ lns, const float* __restrict__ lnb,
                                 _Float16* __restrict__ out, int pairBase, int numPairs)
{
    int wave = threadIdx.x >> 5, lane = threadIdx.x & 31;
    int p = blockIdx.x * 8 + wave;
    if (p >= numPairs) return;
    int gp = pairBase + p;
    int n = gp >> 5, j = gp & 31;
    int nb = nbr[(size_t)n * 32 + j];
    int i = nb < 0 ? 0 : nb;
    int c0 = lane * 2, c1 = c0 + 1;
    float Rn[9], Ri[9];
#pragma unroll
    for (int q = 0; q < 9; q++) { Rn[q] = R[(size_t)n * 9 + q]; Ri[q] = R[(size_t)i * 9 + q]; }
    float can0 = ca[(size_t)n * 3 + 0], can1 = ca[(size_t)n * 3 + 1], can2 = ca[(size_t)n * 3 + 2];
    float cai0 = ca[(size_t)i * 3 + 0], cai1 = ca[(size_t)i * 3 + 1], cai2 = ca[(size_t)i * 3 + 2];
    float a0 = 0.f, a1 = 0.f;
    // relpos one-hot (gather) * same-chain gate
    int rel = resi[i] - resi[n];
    rel = rel < -32 ? -32 : (rel > 32 ? 32 : rel);
    rel += 32;
    if (chain[i] == chain[n] && batch[i] == batch[n]) {
        a0 += Wrel[rel * 64 + c0]; a1 += Wrel[rel * 64 + c1];
    }
    // RBF distance
    float dx = cai0 - can0, dy = cai1 - can1, dz = cai2 - can2;
    float d = sqrtf(dx * dx + dy * dy + dz * dz + 1e-12f);
    const float invsig = 16.0f / 22.0f;
#pragma unroll
    for (int r = 0; r < 16; r++) {
        float u = (d - (22.0f / 15.0f) * r) * invsig;
        float w = __expf(-u * u);
        a0 += w * Wdist[r * 64 + c0]; a1 += w * Wdist[r * 64 + c1];
    }
    // direction in local frame
    float iv = 1.0f / (d + 1e-8f);
    float dv0 = dx * iv, dv1 = dy * iv, dv2 = dz * iv;
#pragma unroll
    for (int dd = 0; dd < 3; dd++) {
        float s = dv0 * Rn[dd] + dv1 * Rn[3 + dd] + dv2 * Rn[6 + dd];
        a0 += s * Wdir[dd * 64 + c0]; a1 += s * Wdir[dd * 64 + c1];
    }
    // relative rotation R_n^T R_i
#pragma unroll
    for (int e = 0; e < 3; e++) {
#pragma unroll
        for (int dd = 0; dd < 3; dd++) {
            float s = Rn[e] * Ri[dd] + Rn[3 + e] * Ri[3 + dd] + Rn[6 + e] * Ri[6 + dd];
            int q = e * 3 + dd;
            a0 += s * Wrot[q * 64 + c0]; a1 += s * Wrot[q * 64 + c1];
        }
    }
    // neighbour atom positions in local frame
    const float* pip = pos + (size_t)i * 42;
    for (int a = 0; a < 14; a++) {
        float px = pip[a * 3 + 0] - can0, py = pip[a * 3 + 1] - can1, pz = pip[a * 3 + 2] - can2;
#pragma unroll
        for (int dd = 0; dd < 3; dd++) {
            float s = px * Rn[dd] + py * Rn[3 + dd] + pz * Rn[6 + dd];
            int q = a * 3 + dd;
            a0 += s * Wpvec[q * 64 + c0]; a1 += s * Wpvec[q * 64 + c1];
        }
    }
    // + aaf[idx]
    a0 += aaf[(size_t)i * 64 + c0]; a1 += aaf[(size_t)i * 64 + c1];
    // LayerNorm over 64 (2 per lane, butterfly over 32 lanes)
    float sum = wsum(a0 + a1);
    float sq  = wsum(a0 * a0 + a1 * a1);
    float mu = sum * (1.0f / 64.0f);
    float var = sq * (1.0f / 64.0f) - mu * mu;
    float rs = rsqrtf(var + 1e-5f);
    out[(size_t)p * 64 + c0] = (_Float16)((a0 - mu) * rs * lns[c0] + lnb[c0]);
    out[(size_t)p * 64 + c1] = (_Float16)((a1 - mu) * rs * lns[c1] + lnb[c1]);
}

// ---------------------------------------------------------------------------
// Attention: one wave per node. Lane l owns head-dim d=l for all 8 heads.
// ---------------------------------------------------------------------------
__launch_bounds__(256)
__global__ void attn_kernel(const float* __restrict__ qkv, const int* __restrict__ nbr,
                            const float* __restrict__ mask, const float* __restrict__ pairBias,
                            _Float16* __restrict__ oF16, int Nn)
{
    int wave = threadIdx.x >> 5, lane = threadIdx.x & 31;
    int n = blockIdx.x * 8 + wave;
    if (n >= Nn) return;
    float q[8];
    const float* qp = qkv + (size_t)n * 768;
#pragma unroll
    for (int h = 0; h < 8; h++) q[h] = qp[h * 32 + lane];
    float mn = mask[n];
    float lj[8];
#pragma unroll
    for (int h = 0; h < 8; h++) lj[h] = -30000.0f;
    const float scale = 0.17677669529663687f; // 1/sqrt(32)
    for (int j = 0; j < 32; j++) {
        int nb = nbr[(size_t)n * 32 + j];
        int i = nb < 0 ? 0 : nb;
        float pm = mn * mask[i] * (nb != -1 ? 1.0f : 0.0f);
        const float* kp = qkv + (size_t)i * 768 + 256;
#pragma unroll
        for (int h = 0; h < 8; h++) {
            float dot = wsum(q[h] * kp[h * 32 + lane]);
            float v = dot * scale + pairBias[((size_t)n * 32 + j) * 16 + h];
            if (!(pm > 0.0f)) v = -30000.0f;
            if (lane == j) lj[h] = v;   // lane j keeps neighbour j's logits
        }
    }
    // softmax over neighbours (lanes)
    float a[8];
#pragma unroll
    for (int h = 0; h < 8; h++) {
        float m = wmaxf(lj[h]);
        float e = __expf(lj[h] - m);
        float s = wsum(e);
        a[h] = e / s;
    }
    float o[8] = {0.f, 0.f, 0.f, 0.f, 0.f, 0.f, 0.f, 0.f};
    for (int j = 0; j < 32; j++) {
        int nb = nbr[(size_t)n * 32 + j];
        int i = nb < 0 ? 0 : nb;
        const float* vp = qkv + (size_t)i * 768 + 512;
#pragma unroll
        for (int h = 0; h < 8; h++) {
            float ah = __shfl(a[h], j, 32);
            o[h] += ah * vp[h * 32 + lane];
        }
    }
#pragma unroll
    for (int h = 0; h < 8; h++) oF16[(size_t)n * 256 + h * 32 + lane] = (_Float16)o[h];
}

// ---------------------------------------------------------------------------
// local1 = LN(features_local + update); residual1 = features_residual + update
// ---------------------------------------------------------------------------
__launch_bounds__(256)
__global__ void ln_res_kernel(const float* __restrict__ fl, const float* __restrict__ fr,
                              const float* __restrict__ upd,
                              const float* __restrict__ s, const float* __restrict__ b,
                              float* __restrict__ outLocal, float* __restrict__ outRes, int Nn)
{
    int wave = threadIdx.x >> 5, lane = threadIdx.x & 31;
    int n = blockIdx.x * 8 + wave;
    if (n >= Nn) return;
    size_t base = (size_t)n * 256 + lane * 8;
    float v[8], u[8], sum = 0.f, sq = 0.f;
#pragma unroll
    for (int r = 0; r < 8; r++) {
        u[r] = upd[base + r];
        v[r] = fl[base + r] + u[r];
        sum += v[r]; sq += v[r] * v[r];
    }
    sum = wsum(sum); sq = wsum(sq);
    float mu = sum * (1.0f / 256.0f);
    float var = sq * (1.0f / 256.0f) - mu * mu;
    float rs = rsqrtf(var + 1e-5f);
#pragma unroll
    for (int r = 0; r < 8; r++) {
        int c = lane * 8 + r;
        outLocal[base + r] = (v[r] - mu) * rs * s[c] + b[c];
        outRes[base + r] = fr[base + r] + u[r];
    }
}

// final: out[0:N*256]=LN(local1+update2,ln2), out[N*256:]=residual1+update2
__launch_bounds__(256)
__global__ void final_kernel(const float* __restrict__ local1, const float* __restrict__ resid1,
                             const float* __restrict__ upd2,
                             const float* __restrict__ s, const float* __restrict__ b,
                             float* __restrict__ out, int Nn)
{
    int wave = threadIdx.x >> 5, lane = threadIdx.x & 31;
    int n = blockIdx.x * 8 + wave;
    if (n >= Nn) return;
    size_t base = (size_t)n * 256 + lane * 8;
    float v[8], u[8], sum = 0.f, sq = 0.f;
#pragma unroll
    for (int r = 0; r < 8; r++) {
        u[r] = upd2[base + r];
        v[r] = local1[base + r] + u[r];
        sum += v[r]; sq += v[r] * v[r];
    }
    sum = wsum(sum); sq = wsum(sq);
    float mu = sum * (1.0f / 256.0f);
    float var = sq * (1.0f / 256.0f) - mu * mu;
    float rs = rsqrtf(var + 1e-5f);
#pragma unroll
    for (int r = 0; r < 8; r++) {
        int c = lane * 8 + r;
        out[base + r] = (v[r] - mu) * rs * s[c] + b[c];
        out[(size_t)Nn * 256 + base + r] = resid1[base + r] + u[r];
    }
}

// g = gelu(x@Wg) * (x@Wu), from concatenated [N x 1024] gemm output -> f16 [N x 512]
__global__ void gate_combine_kernel(const float* __restrict__ go, _Float16* __restrict__ g16, int count)
{
    int t = blockIdx.x * 256 + threadIdx.x;
    if (t >= count) return;
    int n = t >> 9, i = t & 511;
    float a = go[(size_t)n * 1024 + i];         // Wg half
    float u = go[(size_t)n * 1024 + 512 + i];   // Wu half
    g16[t] = (_Float16)(geluf(a) * u);
}

// ---------------------------------------------------------------------------
extern "C" void kernel_launch(void* const* d_in, const int* in_sizes, int n_in,
                              void* d_out, int out_size, void* d_ws, size_t ws_size,
                              hipStream_t stream)
{
    (void)in_sizes; (void)n_in; (void)out_size; (void)ws_size;
    const int*   aa      = (const int*)d_in[0];
    const float* pos     = (const float*)d_in[1];
    const int*   nbr     = (const int*)d_in[2];
    const int*   resi    = (const int*)d_in[3];
    const int*   chain   = (const int*)d_in[4];
    const int*   batch   = (const int*)d_in[5];
    const float* mask    = (const float*)d_in[6];
    const float* flocal  = (const float*)d_in[7];
    const float* fresid  = (const float*)d_in[8];
    const float* W_relpos= (const float*)d_in[9];
    const float* W_dist  = (const float*)d_in[10];
    const float* W_dir   = (const float*)d_in[11];
    const float* W_rot   = (const float*)d_in[12];
    const float* W_pvec  = (const float*)d_in[13];
    const float* W_aa    = (const float*)d_in[14];
    const float* ln_aa_s = (const float*)d_in[15];
    const float* ln_aa_b = (const float*)d_in[16];
    const float* ln_pair_s=(const float*)d_in[17];
    const float* ln_pair_b=(const float*)d_in[18];
    const float* mlp_W1  = (const float*)d_in[19];
    const float* mlp_b1  = (const float*)d_in[20];
    const float* mlp_W2  = (const float*)d_in[21];
    const float* mlp_b2  = (const float*)d_in[22];
    const float* Wq      = (const float*)d_in[23];
    const float* Wk      = (const float*)d_in[24];
    const float* Wv      = (const float*)d_in[25];
    const float* Wb      = (const float*)d_in[26];
    const float* Wo      = (const float*)d_in[27];
    const float* bo      = (const float*)d_in[28];
    const float* ln1_s   = (const float*)d_in[29];
    const float* ln1_b   = (const float*)d_in[30];
    const float* ln2_s   = (const float*)d_in[31];
    const float* ln2_b   = (const float*)d_in[32];
    const float* enc_W1  = (const float*)d_in[33];
    const float* enc_b1  = (const float*)d_in[34];
    const float* enc_W2  = (const float*)d_in[35];
    const float* enc_b2  = (const float*)d_in[36];
    const float* Wu      = (const float*)d_in[37];
    const float* Wg      = (const float*)d_in[38];
    const float* Wout    = (const float*)d_in[39];
    const float* bout    = (const float*)d_in[40];

    const int Nn = NRES;
    const size_t NK = (size_t)Nn * KNBR;       // 524288 pairs
    const int CH = 65536, NCH = (int)(NK / CH);

    // ----- workspace layout (persistent block, then two overlay regions) -----
    size_t off = 0;
    auto alloc = [&](size_t bytes) -> char* {
        char* p = (char*)d_ws + off;
        off += (bytes + 255) & ~(size_t)255;
        return p;
    };
    _Float16* W_qkvT = (_Float16*)alloc((size_t)768 * 256 * 2);
    _Float16* W_oT   = (_Float16*)alloc((size_t)256 * 256 * 2);
    _Float16* W_mlp1T= (_Float16*)alloc((size_t)128 * 64 * 2);
    _Float16* W_cT   = (_Float16*)alloc((size_t)16 * 128 * 2);
    _Float16* W_enc1T= (_Float16*)alloc((size_t)512 * 64 * 2);
    _Float16* W_enc2T= (_Float16*)alloc((size_t)256 * 512 * 2);
    _Float16* W_guT  = (_Float16*)alloc((size_t)1024 * 256 * 2);
    _Float16* W_outT = (_Float16*)alloc((size_t)256 * 512 * 2);
    float* bc        = (float*)alloc(16 * 4);
    float* Rbuf      = (float*)alloc((size_t)Nn * 9 * 4);
    float* cabuf     = (float*)alloc((size_t)Nn * 3 * 4);
    float* aafbuf    = (float*)alloc((size_t)Nn * 64 * 4);
    _Float16* lpF16  = (_Float16*)alloc((size_t)Nn * 64 * 2);
    float* local1    = (float*)alloc((size_t)Nn * 256 * 4);
    float* resid1    = (float*)alloc((size_t)Nn * 256 * 4);
    float* update2   = (float*)alloc((size_t)Nn * 256 * 4);
    // Region A: phase1 (qkv/attention) overlaid by phase2 (post-attention MLPs)
    size_t regA = off;
    _Float16* localF16 = (_Float16*)((char*)d_ws + regA);                              //  8.4 MB
    float* qkv         = (float*)((char*)d_ws + regA + 8388608);                       // 50.3 MB
    float* pairBias    = (float*)((char*)d_ws + regA + 8388608 + 50331648);            // 33.6 MB
    float* gatedOut    = (float*)((char*)d_ws + regA);                                 // 67.1 MB
    _Float16* encHid   = (_Float16*)((char*)d_ws + regA + 67108864);                   // 16.8 MB
    _Float16* xF16     = (_Float16*)((char*)d_ws + regA + 67108864 + 16777216);        //  8.4 MB
    _Float16* gF16     = (_Float16*)((char*)d_ws + regA + 67108864 + 16777216 + 8388608); // 16.8 MB
    size_t regB = regA + 109051904;
    // Region B: pair chunk staging overlaid by attention-output staging
    _Float16* pairChunkF16 = (_Float16*)((char*)d_ws + regB);                          //  8.4 MB
    _Float16* pairHidF16   = (_Float16*)((char*)d_ws + regB + 8388608);                // 16.8 MB
    _Float16* oF16         = (_Float16*)((char*)d_ws + regB);
    float* updBuf          = (float*)((char*)d_ws + regB + 8388608);

    // NT=4 strip GEMM when cols % 64 == 0, else single-tile version
    auto gemm = [&](const _Float16* A, const _Float16* BT, const float* bias,
                    const float* add, float* o32, _Float16* o16,
                    int rows, int kd, int cols, int gelu) {
        if ((cols & 63) == 0) {
            int strips = (rows >> 4) * (cols >> 6);
            int blocks = (strips + 7) >> 3;
            gemm_wmma_kernel<4><<<blocks, 256, 0, stream>>>(A, BT, bias, add, o32, o16,
                                                            rows, kd, cols, gelu);
        } else {
            int tiles = (rows >> 4) * (cols >> 4);
            int blocks = (tiles + 7) >> 3;
            gemm_wmma_kernel<1><<<blocks, 256, 0, stream>>>(A, BT, bias, add, o32, o16,
                                                            rows, kd, cols, gelu);
        }
    };
    auto tr = [&](const float* W, _Float16* WT, int Kd, int Nc, int Kpad) {
        int cnt = Nc * Kpad;
        transpose_f16_kernel<<<(cnt + 255) / 256, 256, 0, stream>>>(W, WT, Kd, Nc, Kpad);
    };

    // ----- weight prep -----
    tr(Wq, W_qkvT, 256, 256, 256);
    tr(Wk, W_qkvT + 256 * 256, 256, 256, 256);
    tr(Wv, W_qkvT + 512 * 256, 256, 256, 256);
    tr(Wo, W_oT, 256, 256, 256);
    tr(mlp_W1, W_mlp1T, 64, 128, 64);
    tr(enc_W1, W_enc1T, 42, 512, 64);
    tr(enc_W2, W_enc2T, 512, 256, 512);
    tr(Wg, W_guT, 256, 512, 256);
    tr(Wu, W_guT + 512 * 256, 256, 512, 256);
    tr(Wout, W_outT, 512, 256, 512);
    make_wc_kernel<<<8, 256, 0, stream>>>(mlp_W2, Wb, mlp_b2, W_cT, bc);

    // ----- frames, aa features, local f16 -----
    frames_kernel<<<Nn / 256, 256, 0, stream>>>(pos, aa, W_aa, ln_aa_s, ln_aa_b,
                                                Rbuf, cabuf, lpF16, aafbuf, Nn);
    f32_to_f16_kernel<<<(Nn * 256) / 256, 256, 0, stream>>>(flocal, localF16, Nn * 256);

    // ----- QKV (single fused WMMA GEMM, cols = 768) -----
    gemm(localF16, W_qkvT, nullptr, nullptr, qkv, nullptr, Nn, 256, 768, 0);

    // ----- pair pipeline in chunks: features+LN -> 64->128 gelu -> @Wc (+bc) -----
    for (int chk = 0; chk < NCH; chk++) {
        int base = chk * CH;
        pair_feat_kernel<<<CH / 8, 256, 0, stream>>>(pos, nbr, resi, chain, batch,
                                                     Rbuf, cabuf, aafbuf,
                                                     W_relpos, W_dist, W_dir, W_rot, W_pvec,
                                                     ln_pair_s, ln_pair_b,
                                                     pairChunkF16, base, CH);
        gemm(pairChunkF16, W_mlp1T, mlp_b1, nullptr, nullptr, pairHidF16, CH, 64, 128, 1);
        gemm(pairHidF16, W_cT, bc, nullptr, pairBias + (size_t)base * 16, nullptr, CH, 128, 16, 0);
    }

    // ----- attention -----
    attn_kernel<<<Nn / 8, 256, 0, stream>>>(qkv, nbr, mask, pairBias, oF16, Nn);

    // ----- update = o @ Wo + bo; local1 = LN(fl + update); residual1 = fr + update -----
    gemm(oF16, W_oT, bo, nullptr, updBuf, nullptr, Nn, 256, 256, 0);
    ln_res_kernel<<<Nn / 8, 256, 0, stream>>>(flocal, fresid, updBuf, ln1_s, ln1_b,
                                              local1, resid1, Nn);

    // ----- encoder: x = local1 + gelu(lp@W1+b1)@W2 + b2 -----
    gemm(lpF16, W_enc1T, enc_b1, nullptr, nullptr, encHid, Nn, 64, 512, 1);
    gemm(encHid, W_enc2T, enc_b2, local1, nullptr, xF16, Nn, 512, 256, 0);

    // ----- gated MLP: update2 = (gelu(x@Wg) * (x@Wu)) @ Wout + bout -----
    gemm(xF16, W_guT, nullptr, nullptr, gatedOut, nullptr, Nn, 256, 1024, 0);
    gate_combine_kernel<<<(Nn * 512) / 256, 256, 0, stream>>>(gatedOut, gF16, Nn * 512);
    gemm(gF16, W_outT, bout, nullptr, update2, nullptr, Nn, 512, 256, 0);

    // ----- final LN + residual -----
    final_kernel<<<Nn / 8, 256, 0, stream>>>(local1, resid1, update2, ln2_s, ln2_b,
                                             (float*)d_out, Nn);
}
